// LineVectorizer_41867341201699
// MI455X (gfx1250) — matrix operand order repaired and checked by reading, not compiled, not run on hardware
//
#include <hip/hip_runtime.h>

typedef __attribute__((ext_vector_type(16))) _Float16 v16h;
typedef __attribute__((ext_vector_type(8)))  float    v8f;

#define HW_    128
#define PIX    16384
#define CIN    256
#define DLOI   128
#define KJ     128
#define NPAIR  8128
#define BATCH  4
#define MTOT   (BATCH*NPAIR)   /* 32512 */
#define DFC    1024
#define KPAD1  1056            /* 1032 padded to multiple of 32 */

// ---------- WMMA 16-bit fragment K index maps (per CDNA5 ISA 7.12.2) ----------
__device__ __forceinline__ int kmapA(int half, int vp) {
  // A 16x32: lanes0-15: V0..3 -> K0..7, V4..7 -> K16..23 ; lanes16-31: +8
  int k = (vp < 4) ? (vp * 2) : (16 + (vp - 4) * 2);
  return k + half * 8;
}
__device__ __forceinline__ int kmapB(int half, int vp) {
  // B 32x16: same K order as A: lanes0-15: V0..3 -> K0..7, V4..7 -> K16..23 ; lanes16-31: +8
  return ((vp < 4) ? (vp * 2) : (16 + (vp - 4) * 2)) + half * 8;
}

// ---------- CDNA5 async global->LDS copy (ASYNCcnt path) ----------
__device__ __forceinline__ unsigned lds_off_u32(const void* p) {
  // generic pointer to LDS: low 32 bits are the wave-relative LDS byte address
  return (unsigned)(unsigned long long)p;
}
__device__ __forceinline__ void async_copy_b128(unsigned lds_off, const void* gaddr) {
  asm volatile("global_load_async_to_lds_b128 %0, %1, off"
               :
               : "v"(lds_off), "v"(gaddr)
               : "memory");
}
__device__ __forceinline__ void wait_async0() {
  asm volatile("s_wait_asynccnt 0x0" ::: "memory");
}

// =====================================================================
// Weight prep: Wt[n*Kpad + k] = (k < Kin) ? (f16)W[k*N + n] : 0
// =====================================================================
__global__ void k_prep_w(const float* __restrict__ W, _Float16* __restrict__ Wt,
                         int N, int Kin, int Kpad) {
  int idx = blockIdx.x * blockDim.x + threadIdx.x;
  if (idx >= N * Kpad) return;
  int n = idx / Kpad, k = idx - n * Kpad;
  Wt[idx] = (k < Kin) ? (_Float16)W[(size_t)k * N + n] : (_Float16)0.f;
}

// =====================================================================
// Conv 1x1 via WMMA: x[b][o][p] = sum_c feature[b][c][p]*fc1_w[o][c] + fc1_b[o]
// block 256 (8 waves), tile 128 pixels x 128 out-channels, K-step 32
// =====================================================================
__global__ __launch_bounds__(256) void k_conv(const float* __restrict__ feature,
                                              const float* __restrict__ fc1_w,
                                              const float* __restrict__ fc1_b,
                                              float* __restrict__ x) {
  __shared__ _Float16 As[128][40];
  __shared__ _Float16 Bs[128][40];
  const int tid = threadIdx.x;
  const int lane = tid & 31, w = tid >> 5;
  const int half = lane >> 4, l15 = lane & 15;
  const int b = blockIdx.y;
  const int p0 = blockIdx.x * 128;

  const v8f zf = {0.f,0.f,0.f,0.f,0.f,0.f,0.f,0.f};
  v8f acc[8];
#pragma unroll
  for (int i = 0; i < 8; ++i) acc[i] = zf;

  for (int ks = 0; ks < CIN / 32; ++ks) {
    const int k0 = ks * 32;
    {
      // A tile: As[row][c] = feature[b][k0+c][p0+row]; batched float4 loads
      const int c = tid >> 3, rb = (tid & 7) * 16;
      const float* src = feature + ((size_t)b * CIN + k0 + c) * PIX + p0 + rb;
      float4 f0 = *reinterpret_cast<const float4*>(src);
      float4 f1 = *reinterpret_cast<const float4*>(src + 4);
      float4 f2 = *reinterpret_cast<const float4*>(src + 8);
      float4 f3 = *reinterpret_cast<const float4*>(src + 12);
      float fv[16] = {f0.x,f0.y,f0.z,f0.w, f1.x,f1.y,f1.z,f1.w,
                      f2.x,f2.y,f2.z,f2.w, f3.x,f3.y,f3.z,f3.w};
#pragma unroll
      for (int i = 0; i < 16; ++i) As[rb + i][c] = (_Float16)fv[i];

      // B tile: Bs[n][k] = fc1_w[n][k0+k]; batched loads + packed stores
      const int n = tid >> 1, kq = (tid & 1) * 16;
      const float* sw = fc1_w + (size_t)n * CIN + k0 + kq;
      float4 g0 = *reinterpret_cast<const float4*>(sw);
      float4 g1 = *reinterpret_cast<const float4*>(sw + 4);
      float4 g2 = *reinterpret_cast<const float4*>(sw + 8);
      float4 g3 = *reinterpret_cast<const float4*>(sw + 12);
      float gv[16] = {g0.x,g0.y,g0.z,g0.w, g1.x,g1.y,g1.z,g1.w,
                      g2.x,g2.y,g2.z,g2.w, g3.x,g3.y,g3.z,g3.w};
      _Float16 hv[16];
#pragma unroll
      for (int i = 0; i < 16; ++i) hv[i] = (_Float16)gv[i];
      *reinterpret_cast<uint4*>(&Bs[n][kq])     = *reinterpret_cast<uint4*>(&hv[0]);
      *reinterpret_cast<uint4*>(&Bs[n][kq + 8]) = *reinterpret_cast<uint4*>(&hv[8]);
    }
    __syncthreads();

    const _Float16* Arow = &As[w * 16 + l15][0];
    v16h a;
#pragma unroll
    for (int vp = 0; vp < 8; ++vp) {
      int kk = kmapA(half, vp);
      a[2 * vp]     = Arow[kk];
      a[2 * vp + 1] = Arow[kk + 1];
    }
#pragma unroll
    for (int nf = 0; nf < 8; ++nf) {
      const _Float16* Brow = &Bs[nf * 16 + l15][0];
      v16h bf;
#pragma unroll
      for (int vp = 0; vp < 8; ++vp) {
        int kk = kmapB(half, vp);
        bf[2 * vp]     = Brow[kk];
        bf[2 * vp + 1] = Brow[kk + 1];
      }
      acc[nf] = __builtin_amdgcn_wmma_f32_16x16x32_f16(
          false, a, false, bf, (short)0, acc[nf], false, false);
    }
    __syncthreads();
  }

  float* xb = x + (size_t)b * DLOI * PIX;
#pragma unroll
  for (int nf = 0; nf < 8; ++nf) {
    int n = nf * 16 + l15;
    float bv = fc1_b[n];
    size_t base = (size_t)n * PIX + p0 + w * 16 + half * 8;
    float4 v0 = make_float4(acc[nf][0] + bv, acc[nf][1] + bv, acc[nf][2] + bv, acc[nf][3] + bv);
    float4 v1 = make_float4(acc[nf][4] + bv, acc[nf][5] + bv, acc[nf][6] + bv, acc[nf][7] + bv);
    *(float4*)(xb + base)     = v0;
    *(float4*)(xb + base + 4) = v1;
  }
}

// =====================================================================
// Junction NMS + iterative top-128 per image. block 256, grid B.
// =====================================================================
__global__ __launch_bounds__(256) void k_junc(const float* __restrict__ jmap,
                                              const float* __restrict__ joff,
                                              float* __restrict__ nms,
                                              float* __restrict__ xyw,
                                              float* __restrict__ jsc) {
  const int b = blockIdx.x, tid = threadIdx.x;
  const float* jm = jmap + (size_t)b * PIX;
  float* nm = nms + (size_t)b * PIX;
  for (int i = tid; i < PIX; i += 256) {
    int y = i >> 7, xc = i & 127;
    float v = jm[i], mx = -1e30f;
    for (int dy = -1; dy <= 1; ++dy)
      for (int dx = -1; dx <= 1; ++dx) {
        int yy = y + dy, xx = xc + dx;
        if (yy >= 0 && yy < HW_ && xx >= 0 && xx < HW_) {
          float t = jm[yy * HW_ + xx];
          mx = t > mx ? t : mx;
        }
      }
    nm[i] = (v == mx) ? v : 0.f;
  }
  __threadfence_block();
  __syncthreads();

  __shared__ float bval[256];
  __shared__ int bidx[256];
  for (int kk = 0; kk < KJ; ++kk) {
    float best = -1e30f; int bi = 0x7fffffff;
    for (int i = tid; i < PIX; i += 256) {
      float v = nm[i];
      if (v > best || (v == best && i < bi)) { best = v; bi = i; }
    }
    bval[tid] = best; bidx[tid] = bi;
    __syncthreads();
    for (int off = 128; off > 0; off >>= 1) {
      if (tid < off) {
        if (bval[tid + off] > bval[tid] ||
            (bval[tid + off] == bval[tid] && bidx[tid + off] < bidx[tid])) {
          bval[tid] = bval[tid + off]; bidx[tid] = bidx[tid + off];
        }
      }
      __syncthreads();
    }
    if (tid == 0) {
      int idx = bidx[0];
      jsc[b * KJ + kk] = bval[0];
      float yy = (float)(idx >> 7) + joff[(size_t)b * 2 * PIX + idx] + 0.5f;
      float xx = (float)(idx & 127) + joff[(size_t)b * 2 * PIX + PIX + idx] + 0.5f;
      xyw[((size_t)b * KJ + kk) * 2 + 0] = yy;
      xyw[((size_t)b * KJ + kk) * 2 + 1] = xx;
      nm[idx] = -1e30f;
      __threadfence_block();
    }
    __syncthreads();
  }
}

// =====================================================================
// Row building: per (pair, image) block of 128 threads (one per LOI channel).
// =====================================================================
__global__ __launch_bounds__(128) void k_rows(const float* __restrict__ x,
                                              const float* __restrict__ xyw,
                                              _Float16* __restrict__ rows16,
                                              float* __restrict__ lines) {
  const int pair = blockIdx.x, b = blockIdx.y, tid = threadIdx.x;
  __shared__ int si0[32], sj0[32], si1[32], sj1[32];
  __shared__ float w00[32], w10[32], w01[32], w11[32];
  __shared__ float pf[8];

  int r = pair, u = 0;
  while (r >= (KJ - 1) - u) { r -= (KJ - 1) - u; ++u; }
  int v = u + 1 + r;

  const float* xyb = xyw + (size_t)b * KJ * 2;
  float uy = xyb[u * 2], ux = xyb[u * 2 + 1];
  float vy = xyb[v * 2], vx = xyb[v * 2 + 1];

  if (tid < 32) {
    float lam = (float)tid / 31.0f;
    float px = uy * lam + vy * (1.f - lam) - 0.5f;
    float py = ux * lam + vx * (1.f - lam) - 0.5f;
    float px0 = fminf(fmaxf(floorf(px), 0.f), 127.f);
    float py0 = fminf(fmaxf(floorf(py), 0.f), 127.f);
    float px1 = fminf(px0 + 1.f, 127.f);
    float py1 = fminf(py0 + 1.f, 127.f);
    si0[tid] = (int)px0; sj0[tid] = (int)py0;
    si1[tid] = (int)px1; sj1[tid] = (int)py1;
    w00[tid] = (px1 - px) * (py1 - py);
    w10[tid] = (px - px0) * (py1 - py);
    w01[tid] = (px1 - px) * (py - py0);
    w11[tid] = (px - px0) * (py - py0);
  }
  if (tid == 0) {
    float dy = uy - vy, dx = ux - vx;
    float nrm = sqrtf(dy * dy + dx * dx);
    nrm = nrm < 1e-6f ? 1e-6f : nrm;
    pf[0] = uy / (float)HW_; pf[1] = ux / (float)HW_;
    pf[2] = vy / (float)HW_; pf[3] = vx / (float)HW_;
    pf[4] = dy / nrm;        pf[5] = dx / nrm;
    pf[6] = (u > KJ) ? 1.f : 0.f; pf[7] = (v > KJ) ? 1.f : 0.f;
    float* lo = lines + ((size_t)b * NPAIR + pair) * 4;
    lo[0] = uy; lo[1] = ux; lo[2] = vy; lo[3] = vx;
  }
  __syncthreads();

  const size_t m = (size_t)b * NPAIR + pair;
  const float* xc = x + ((size_t)b * DLOI + tid) * PIX;
  _Float16* rowp = rows16 + m * KPAD1;
#pragma unroll
  for (int sg = 0; sg < 8; ++sg) {
    float mx = -1e30f;
#pragma unroll
    for (int q = 0; q < 4; ++q) {
      int t = sg * 4 + q;
      float val = xc[si0[t] * HW_ + sj0[t]] * w00[t]
                + xc[si1[t] * HW_ + sj0[t]] * w10[t]
                + xc[si0[t] * HW_ + sj1[t]] * w01[t]
                + xc[si1[t] * HW_ + sj1[t]] * w11[t];
      mx = fmaxf(mx, val);
    }
    rowp[tid * 8 + sg] = (_Float16)mx;
  }
  if (tid < 32) rowp[1024 + tid] = (tid < 8) ? (_Float16)pf[tid] : (_Float16)0.f;
}

// =====================================================================
// f16 GEMM + bias + relu -> f16, with CDNA5 async global->LDS staging.
// A: MxKpad row-major; Bt: NxKpad row-major. block 256, tile 128x128, K-step 32.
// =====================================================================
__global__ __launch_bounds__(256) void k_gemm_f16(const _Float16* __restrict__ A, int lda,
                                                  const _Float16* __restrict__ Bt, int ldb,
                                                  const float* __restrict__ bias,
                                                  _Float16* __restrict__ C, int ldc,
                                                  int ksteps) {
  __shared__ _Float16 As[128][40];
  __shared__ _Float16 Bs[128][40];
  const int tid = threadIdx.x;
  const int lane = tid & 31, w = tid >> 5;
  const int half = lane >> 4, l15 = lane & 15;
  const int m0 = blockIdx.x * 128;
  const int n0 = blockIdx.y * 128;

  const v8f zf = {0.f,0.f,0.f,0.f,0.f,0.f,0.f,0.f};
  v8f acc[8];
#pragma unroll
  for (int i = 0; i < 8; ++i) acc[i] = zf;

  const int row = tid >> 1, col0 = (tid & 1) * 16;
  const unsigned ldsA = lds_off_u32(&As[row][col0]);
  const unsigned ldsB = lds_off_u32(&Bs[row][col0]);

  for (int ks = 0; ks < ksteps; ++ks) {
    const int k0 = ks * 32;
    {
      const _Float16* srcA = A + (size_t)(m0 + row) * lda + k0 + col0;
      const _Float16* srcB = Bt + (size_t)(n0 + row) * ldb + k0 + col0;
      // async global->LDS (tracked by ASYNCcnt)
      async_copy_b128(ldsA,      srcA);
      async_copy_b128(ldsA + 16, srcA + 8);
      async_copy_b128(ldsB,      srcB);
      async_copy_b128(ldsB + 16, srcB + 8);
      if (ks + 1 < ksteps) {  // gfx1250 global_prefetch_b8
        __builtin_prefetch(srcA + 32, 0, 1);
        __builtin_prefetch(srcB + 32, 0, 1);
      }
    }
    wait_async0();
    __syncthreads();

    const _Float16* Arow = &As[w * 16 + l15][0];
    v16h a;
#pragma unroll
    for (int vp = 0; vp < 8; ++vp) {
      int kk = kmapA(half, vp);
      a[2 * vp]     = Arow[kk];
      a[2 * vp + 1] = Arow[kk + 1];
    }
#pragma unroll
    for (int nf = 0; nf < 8; ++nf) {
      const _Float16* Brow = &Bs[nf * 16 + l15][0];
      v16h bf;
#pragma unroll
      for (int vp = 0; vp < 8; ++vp) {
        int kk = kmapB(half, vp);
        bf[2 * vp]     = Brow[kk];
        bf[2 * vp + 1] = Brow[kk + 1];
      }
      acc[nf] = __builtin_amdgcn_wmma_f32_16x16x32_f16(
          false, a, false, bf, (short)0, acc[nf], false, false);
    }
    __syncthreads();
  }

#pragma unroll
  for (int nf = 0; nf < 8; ++nf) {
    int n = n0 + nf * 16 + l15;
    float bv = bias[n];
#pragma unroll
    for (int rr = 0; rr < 8; ++rr) {
      int mm = m0 + w * 16 + half * 8 + rr;
      float vv = acc[nf][rr] + bv;
      vv = vv > 0.f ? vv : 0.f;
      C[(size_t)mm * ldc + n] = (_Float16)vv;
    }
  }
}

// =====================================================================
// Final dot with W3 + sigmoid. one wave per row.
// =====================================================================
__global__ __launch_bounds__(256) void k_score(const _Float16* __restrict__ h2,
                                               const float* __restrict__ W3,
                                               const float* __restrict__ b3,
                                               float* __restrict__ s) {
  int wid = blockIdx.x * (blockDim.x >> 5) + (threadIdx.x >> 5);
  int lane = threadIdx.x & 31;
  if (wid >= MTOT) return;
  const _Float16* row = h2 + (size_t)wid * DFC;
  float acc = 0.f;
  for (int j = lane; j < DFC; j += 32) acc += (float)row[j] * W3[j];
  for (int off = 16; off > 0; off >>= 1) acc += __shfl_xor(acc, off, 32);
  if (lane == 0) s[wid] = 1.f / (1.f + expf(-(acc + b3[0])));
}

// =====================================================================
// Post: per-image bitonic sort (8192, desc) of masked scores in LDS,
// select top lines + junctions, write outputs.
// =====================================================================
__global__ __launch_bounds__(1024) void k_post(const float* __restrict__ s,
                                               const float* __restrict__ lines,
                                               const float* __restrict__ xyw,
                                               const float* __restrict__ jsc,
                                               float* __restrict__ out,
                                               int* __restrict__ cnt_ws) {
  __shared__ float key[8192];
  __shared__ int pay[8192];
  const int b = blockIdx.x, tid = threadIdx.x;
  const float* sb = s + (size_t)b * NPAIR;

  for (int i = tid; i < 8192; i += 1024) {
    float kv = -2.0f;
    if (i < NPAIR) { float v = sb[i]; kv = (v > 0.5f) ? v : -1.0f; }
    key[i] = kv; pay[i] = i;
  }
  __syncthreads();

  for (int k = 2; k <= 8192; k <<= 1) {
    for (int j = k >> 1; j > 0; j >>= 1) {
      for (int i = tid; i < 8192; i += 1024) {
        int ixj = i ^ j;
        if (ixj > i) {
          bool desc = ((i & k) == 0);
          if ((key[i] < key[ixj]) == desc) {
            float tk = key[i]; key[i] = key[ixj]; key[ixj] = tk;
            int   tp = pay[i]; pay[i] = pay[ixj]; pay[ixj] = tp;
          }
        }
      }
      __syncthreads();
    }
  }

  volatile int* vc = cnt_ws + b * 2;
  for (int i = tid; i < 8192; i += 1024) {
    bool v = key[i] > 0.5f;
    bool vn = (i + 1 < 8192) ? (key[i + 1] > 0.5f) : false;
    if (v && !vn) vc[0] = i + 1;
    if (i == 0 && !v) vc[0] = 0;
  }
  const float* js = jsc + (size_t)b * KJ;
  if (tid < KJ) {
    bool v = js[tid] > 0.03f;
    bool vn = (tid + 1 < KJ) ? (js[tid + 1] > 0.03f) : false;
    if (v && !vn) vc[1] = tid + 1;
    if (tid == 0 && !v) vc[1] = 0;
  }
  __threadfence();
  __syncthreads();
  const int cnt = vc[0], cj = vc[1];
  const int cm  = cnt > 0 ? cnt : 1;
  const int cjm = cj  > 0 ? cj  : 1;

  const float* lb = lines + (size_t)b * NPAIR * 4;
  float* out_lines = out + (size_t)b * 10000;                 // (2500,2,2)
  float* out_score = out + 40000 + (size_t)b * 2500;
  float* out_junc  = out + 50000 + (size_t)b * 500;           // (250,2)

  for (int i = tid; i < 2500; i += 1024) {
    int sel = pay[i % cm];
    if (cnt > 0) {
      out_lines[i * 4 + 0] = lb[sel * 4 + 0];
      out_lines[i * 4 + 1] = lb[sel * 4 + 1];
      out_lines[i * 4 + 2] = lb[sel * 4 + 2];
      out_lines[i * 4 + 3] = lb[sel * 4 + 3];
      out_score[i] = sb[sel];
    } else {
      out_lines[i * 4 + 0] = 0.f; out_lines[i * 4 + 1] = 0.f;
      out_lines[i * 4 + 2] = 0.f; out_lines[i * 4 + 3] = 0.f;
      out_score[i] = 0.f;
    }
  }
  const float* xyb = xyw + (size_t)b * KJ * 2;
  for (int i = tid; i < 250; i += 1024) {
    int ji = i % cjm;
    out_junc[i * 2 + 0] = cj > 0 ? xyb[ji * 2 + 0] : 0.f;
    out_junc[i * 2 + 1] = cj > 0 ? xyb[ji * 2 + 1] : 0.f;
  }
}

// =====================================================================
extern "C" void kernel_launch(void* const* d_in, const int* in_sizes, int n_in,
                              void* d_out, int out_size, void* d_ws, size_t ws_size,
                              hipStream_t stream) {
  (void)in_sizes; (void)n_in; (void)out_size; (void)ws_size;
  const float* feature = (const float*)d_in[0];
  const float* jmap    = (const float*)d_in[1];
  const float* joff    = (const float*)d_in[2];
  const float* fc1_w   = (const float*)d_in[3];
  const float* fc1_b   = (const float*)d_in[4];
  const float* W1      = (const float*)d_in[5];
  const float* b1      = (const float*)d_in[6];
  const float* W2      = (const float*)d_in[7];
  const float* b2      = (const float*)d_in[8];
  const float* W3      = (const float*)d_in[9];
  const float* b3      = (const float*)d_in[10];

  char* p = (char*)d_ws;
  auto alloc = [&](size_t bytes) -> void* {
    void* r = (void*)p;
    p += (bytes + 255) & ~(size_t)255;
    return r;
  };
  float*    x      = (float*)   alloc((size_t)BATCH * DLOI * PIX * 4);
  _Float16* rows16 = (_Float16*)alloc((size_t)MTOT * KPAD1 * 2);
  _Float16* h1     = (_Float16*)alloc((size_t)MTOT * DFC * 2);
  _Float16* h2     = (_Float16*)alloc((size_t)MTOT * DFC * 2);
  _Float16* w1t    = (_Float16*)alloc((size_t)DFC * KPAD1 * 2);
  _Float16* w2t    = (_Float16*)alloc((size_t)DFC * DFC * 2);
  float*    sbuf   = (float*)   alloc((size_t)MTOT * 4);
  float*    lines  = (float*)   alloc((size_t)BATCH * NPAIR * 4 * 4);
  float*    xyw    = (float*)   alloc((size_t)BATCH * KJ * 2 * 4);
  float*    jsc    = (float*)   alloc((size_t)BATCH * KJ * 4);
  float*    nms    = (float*)   alloc((size_t)BATCH * PIX * 4);
  int*      cnts   = (int*)     alloc((size_t)BATCH * 2 * 4);

  k_prep_w<<<(DFC * KPAD1 + 255) / 256, 256, 0, stream>>>(W1, w1t, DFC, 1032, KPAD1);
  k_prep_w<<<(DFC * DFC + 255) / 256, 256, 0, stream>>>(W2, w2t, DFC, DFC, DFC);
  k_conv<<<dim3(PIX / 128, BATCH), 256, 0, stream>>>(feature, fc1_w, fc1_b, x);
  k_junc<<<BATCH, 256, 0, stream>>>(jmap, joff, nms, xyw, jsc);
  k_rows<<<dim3(NPAIR, BATCH), 128, 0, stream>>>(x, xyw, rows16, lines);
  k_gemm_f16<<<dim3(MTOT / 128, DFC / 128), 256, 0, stream>>>(rows16, KPAD1, w1t, KPAD1, b1, h1, DFC, KPAD1 / 32);
  k_gemm_f16<<<dim3(MTOT / 128, DFC / 128), 256, 0, stream>>>(h1, DFC, w2t, DFC, b2, h2, DFC, DFC / 32);
  k_score<<<MTOT / 8, 256, 0, stream>>>(h2, W3, b3, sbuf);
  k_post<<<BATCH, 1024, 0, stream>>>(sbuf, lines, xyw, jsc, (float*)d_out, cnts);
}
